// BPA_91139206021581
// MI455X (gfx1250) — matrix operand order repaired and checked by reading, not compiled, not run on hardware
//
#include <hip/hip_runtime.h>
#include <hip/hip_bf16.h>

#define N 8192
#define D 2048

typedef __attribute__((ext_vector_type(16))) __bf16 v16bf;
typedef __attribute__((ext_vector_type(8)))  float  v8f;

// ---------------------------------------------------------------------------
// 1) Row L2-normalize; emit hi/lo bf16 split (hi = bf16(v), lo = bf16(v-hi))
// ---------------------------------------------------------------------------
__global__ void k_normalize(const float* __restrict__ x,
                            __hip_bfloat16* __restrict__ xhi,
                            __hip_bfloat16* __restrict__ xlo) {
    const int row = blockIdx.x;
    const float* xr = x + (size_t)row * D;
    float s = 0.f;
    for (int j = threadIdx.x; j < D; j += 256) { float v = xr[j]; s += v * v; }
    __shared__ float red[256];
    red[threadIdx.x] = s;
    __syncthreads();
    for (int off = 128; off > 0; off >>= 1) {
        if (threadIdx.x < off) red[threadIdx.x] += red[threadIdx.x + off];
        __syncthreads();
    }
    const float inv = 1.0f / fmaxf(sqrtf(red[0]), 1e-12f);
    __hip_bfloat16* oh = xhi + (size_t)row * D;
    __hip_bfloat16* ol = xlo + (size_t)row * D;
    for (int j = threadIdx.x; j < D; j += 256) {
        float v = xr[j] * inv;
        __hip_bfloat16 h = __float2bfloat16(v);
        oh[j] = h;
        ol[j] = __float2bfloat16(v - __bfloat162float(h));
    }
}

// ---------------------------------------------------------------------------
// Async memory->LDS staging of one 128x32 bf16 tile (2 x b128 per thread),
// using the gfx1250 ASYNCcnt path (no VGPR staging, enables double buffer).
// ---------------------------------------------------------------------------
__device__ __forceinline__ void async_tile(uint32_t ldsBase,
                                           const __hip_bfloat16* __restrict__ gsrc,
                                           int idxBase, int kb, int t) {
#pragma unroll
    for (int q = 0; q < 2; ++q) {
        const int idx = t + q * 256;            // 0..511 uint4 slots
        const int r   = idx >> 2;
        const int ko  = (idx & 3) << 3;
        const uint32_t lds = ldsBase + (uint32_t)idx * 16u;
        const unsigned long long ga =
            (unsigned long long)(uintptr_t)(gsrc + (size_t)(idxBase + r) * D + kb + ko);
        asm volatile("global_load_async_to_lds_b128 %0, %1, off"
                     :: "v"(lds), "v"(ga) : "memory");
    }
}

// ---------------------------------------------------------------------------
// 2) Gram via WMMA:  S = Xn*Xn^T (hi*hi + hi*lo + lo*hi), fused into
//    log_K = 10*(S - 1), diagonal = -1e6, written straight to d_out.
//    Block = 128x128 tile, 8 waves, each wave a 16x128 strip.
//    Double-buffered async-to-LDS pipeline: one s_wait_asynccnt + one barrier
//    per 32-wide K-step; prefetch of step i+1 overlaps the 24 WMMAs of step i.
// ---------------------------------------------------------------------------
__global__ void __launch_bounds__(256) k_gram(const __hip_bfloat16* __restrict__ xhi,
                                              const __hip_bfloat16* __restrict__ xlo,
                                              float* __restrict__ logK) {
    // [buf][tile: Ah,Al,Bh,Bl][128*32 bf16]
    __shared__ __align__(16) __bf16 smem[2][4][128 * 32];

    const int t    = threadIdx.x;
    const int lane = t & 31;
    const int wave = t >> 5;       // 0..7 -> 16-row strip of the tile
    const int half = lane >> 4;    // 0/1
    const int l16  = lane & 15;

    const int colBase = blockIdx.x * 128;
    const int rowBase = blockIdx.y * 128;

    v8f acc[8];
#pragma unroll
    for (int nt = 0; nt < 8; ++nt)
#pragma unroll
        for (int r = 0; r < 8; ++r) acc[nt][r] = 0.0f;

    // LDS aperture address = low 32 bits of the generic pointer (ISA 10.2)
    uint32_t ldsB[2][4];
#pragma unroll
    for (int b = 0; b < 2; ++b)
#pragma unroll
        for (int m = 0; m < 4; ++m)
            ldsB[b][m] = (uint32_t)(uintptr_t)(&smem[b][m][0]);

    // prologue: prefetch K-chunk 0 into buffer 0
    async_tile(ldsB[0][0], xhi, rowBase, 0, t);
    async_tile(ldsB[0][1], xlo, rowBase, 0, t);
    async_tile(ldsB[0][2], xhi, colBase, 0, t);
    async_tile(ldsB[0][3], xlo, colBase, 0, t);

    for (int kb = 0, buf = 0; kb < D; kb += 32, buf ^= 1) {
        asm volatile("s_wait_asynccnt 0" ::: "memory"); // my loads into smem[buf] done
        __syncthreads();                               // all waves' loads published;
                                                       // smem[buf^1] no longer being read
        if (kb + 32 < D) {                             // prefetch next chunk (overlaps WMMA)
            async_tile(ldsB[buf ^ 1][0], xhi, rowBase, kb + 32, t);
            async_tile(ldsB[buf ^ 1][1], xlo, rowBase, kb + 32, t);
            async_tile(ldsB[buf ^ 1][2], xhi, colBase, kb + 32, t);
            async_tile(ldsB[buf ^ 1][3], xlo, colBase, kb + 32, t);
        }

        const __bf16* Ah = smem[buf][0];
        const __bf16* Al = smem[buf][1];
        const __bf16* Bh = smem[buf][2];
        const __bf16* Bl = smem[buf][3];

        // A fragments (16x32 bf16 layout): lane holds row M=l16;
        // K = half*16+e
        v16bf ah, al;
        {
            const __bf16* Arh = &Ah[(wave * 16 + l16) * 32];
            const __bf16* Arl = &Al[(wave * 16 + l16) * 32];
#pragma unroll
            for (int e = 0; e < 16; ++e) {
                const int k = half * 16 + e;
                ah[e] = Arh[k];
                al[e] = Arl[k];
            }
        }
#pragma unroll
        for (int nt = 0; nt < 8; ++nt) {
            // B fragments (32x16 bf16 layout): lane holds col N=l16, K = half*16+e
            v16bf bh, bl;
            const __bf16* Brh = &Bh[(nt * 16 + l16) * 32];
            const __bf16* Brl = &Bl[(nt * 16 + l16) * 32];
#pragma unroll
            for (int e = 0; e < 16; ++e) {
                bh[e] = Brh[half * 16 + e];
                bl[e] = Brl[half * 16 + e];
            }
            acc[nt] = __builtin_amdgcn_wmma_f32_16x16x32_bf16(
                false, ah, false, bh, (short)0, acc[nt], false, false);
            acc[nt] = __builtin_amdgcn_wmma_f32_16x16x32_bf16(
                false, ah, false, bl, (short)0, acc[nt], false, false);
            acc[nt] = __builtin_amdgcn_wmma_f32_16x16x32_bf16(
                false, al, false, bh, (short)0, acc[nt], false, false);
        }
    }

    // C/D layout: VGPR r, lanes<16 -> M=r ; lanes>=16 -> M=r+8 ; N = l16
#pragma unroll
    for (int nt = 0; nt < 8; ++nt) {
        const int gcol = colBase + nt * 16 + l16;
#pragma unroll
        for (int r = 0; r < 8; ++r) {
            const int grow = rowBase + wave * 16 + half * 8 + r;
            float lk = 10.0f * (acc[nt][r] - 1.0f);   // -C/OT_REG
            if (grow == gcol) lk = -1.0e6f;           // -DIAG_VAL/OT_REG
            logK[(size_t)grow * N + gcol] = lk;
        }
    }
}

// ---------------------------------------------------------------------------
// 3) Sinkhorn passes (streaming, fp32-safe without max-shift: args in [-30,12])
// ---------------------------------------------------------------------------
__global__ void k_zero(float* __restrict__ p) {
    p[blockIdx.x * 256 + threadIdx.x] = 0.0f;
}

__global__ void k_rowpass(const float* __restrict__ logK,
                          const float* __restrict__ log_v,
                          float* __restrict__ log_u) {
    const int i = blockIdx.x;
    const float4* Kr = (const float4*)(logK + (size_t)i * N);
    const float4* lv = (const float4*)log_v;
    float s = 0.f;
    for (int j = threadIdx.x; j < N / 4; j += 256) {
        float4 k4 = Kr[j], v4 = lv[j];
        s += __expf(k4.x + v4.x) + __expf(k4.y + v4.y) +
             __expf(k4.z + v4.z) + __expf(k4.w + v4.w);
    }
    __shared__ float red[256];
    red[threadIdx.x] = s; __syncthreads();
    for (int off = 128; off > 0; off >>= 1) {
        if (threadIdx.x < off) red[threadIdx.x] += red[threadIdx.x + off];
        __syncthreads();
    }
    if (threadIdx.x == 0) log_u[i] = -__logf(red[0]);
}

__global__ void k_colpartial(const float* __restrict__ logK,
                             const float* __restrict__ log_u,
                             float* __restrict__ partial) {
    const int j  = blockIdx.x * 256 + threadIdx.x;   // column (coalesced)
    const int i0 = blockIdx.y * 128;                 // row chunk
    __shared__ float lu[128];
    if (threadIdx.x < 128) lu[threadIdx.x] = log_u[i0 + threadIdx.x];
    __syncthreads();
    float s = 0.f;
#pragma unroll 4
    for (int r = 0; r < 128; ++r)
        s += __expf(lu[r] + logK[(size_t)(i0 + r) * N + j]);
    partial[(size_t)blockIdx.y * N + j] = s;
}

__global__ void k_colfinish(const float* __restrict__ partial,
                            float* __restrict__ log_v) {
    const int j = blockIdx.x * 256 + threadIdx.x;
    float s = 0.f;
#pragma unroll
    for (int c = 0; c < 64; ++c) s += partial[(size_t)c * N + j];
    log_v[j] = -__logf(s);
}

// ---------------------------------------------------------------------------
// 4) Global max of log_P (P/P.max() == exp(log_P - max log_P)), then finalize
// ---------------------------------------------------------------------------
__global__ void k_rowmax(const float* __restrict__ logK,
                         const float* __restrict__ log_u,
                         const float* __restrict__ log_v,
                         float* __restrict__ rowmax) {
    const int i = blockIdx.x;
    const float4* Kr = (const float4*)(logK + (size_t)i * N);
    const float4* lv = (const float4*)log_v;
    float m = -3.0e38f;
    for (int j = threadIdx.x; j < N / 4; j += 256) {
        float4 k4 = Kr[j], v4 = lv[j];
        m = fmaxf(m, fmaxf(fmaxf(k4.x + v4.x, k4.y + v4.y),
                           fmaxf(k4.z + v4.z, k4.w + v4.w)));
    }
    __shared__ float red[256];
    red[threadIdx.x] = m; __syncthreads();
    for (int off = 128; off > 0; off >>= 1) {
        if (threadIdx.x < off)
            red[threadIdx.x] = fmaxf(red[threadIdx.x], red[threadIdx.x + off]);
        __syncthreads();
    }
    if (threadIdx.x == 0) rowmax[i] = log_u[i] + red[0];
}

__global__ void k_gmax(const float* __restrict__ rowmax, float* __restrict__ gmax) {
    float m = -3.0e38f;
    for (int i = threadIdx.x; i < N; i += 256) m = fmaxf(m, rowmax[i]);
    __shared__ float red[256];
    red[threadIdx.x] = m; __syncthreads();
    for (int off = 128; off > 0; off >>= 1) {
        if (threadIdx.x < off)
            red[threadIdx.x] = fmaxf(red[threadIdx.x], red[threadIdx.x + off]);
        __syncthreads();
    }
    if (threadIdx.x == 0) *gmax = red[0];
}

__global__ void k_final(float* __restrict__ P,            // logK in, P out (in place)
                        const float* __restrict__ log_u,
                        const float* __restrict__ log_v,
                        const float* __restrict__ gmax) {
    const int i = blockIdx.x;
    const float base = log_u[i] - *gmax;
    float4* row = (float4*)(P + (size_t)i * N);
    const float4* lv = (const float4*)log_v;
    for (int j = threadIdx.x; j < N / 4; j += 256) {
        float4 k4 = row[j], v4 = lv[j], o;
        o.x = __expf(k4.x + v4.x + base);
        o.y = __expf(k4.y + v4.y + base);
        o.z = __expf(k4.z + v4.z + base);
        o.w = __expf(k4.w + v4.w + base);
        const int jc = j * 4;
        if (i >= jc && i < jc + 4) ((float*)&o)[i - jc] = 1.0f;  // diag -> 1
        row[j] = o;
    }
}

// ---------------------------------------------------------------------------
extern "C" void kernel_launch(void* const* d_in, const int* in_sizes, int n_in,
                              void* d_out, int out_size, void* d_ws, size_t ws_size,
                              hipStream_t stream) {
    (void)in_sizes; (void)n_in; (void)out_size; (void)ws_size;
    const float* x = (const float*)d_in[0];
    float* out = (float*)d_out;              // doubles as logK storage, finalized in place
    char* ws = (char*)d_ws;

    __hip_bfloat16* xhi = (__hip_bfloat16*)ws;                        // 32 MB
    __hip_bfloat16* xlo = xhi + (size_t)N * D;                        // 32 MB
    float* log_u   = (float*)(xlo + (size_t)N * D);
    float* log_v   = log_u + N;
    float* partial = log_v + N;                                       // 64*N floats
    float* rowmax  = partial + (size_t)64 * N;
    float* gmax    = rowmax + N;

    k_normalize<<<N, 256, 0, stream>>>(x, xhi, xlo);
    k_gram<<<dim3(N / 128, N / 128), 256, 0, stream>>>(xhi, xlo, out);
    k_zero<<<N / 256, 256, 0, stream>>>(log_v);
    for (int it = 0; it < 10; ++it) {
        k_rowpass<<<N, 256, 0, stream>>>(out, log_v, log_u);
        k_colpartial<<<dim3(N / 256, 64), 256, 0, stream>>>(out, log_u, partial);
        k_colfinish<<<N / 256, 256, 0, stream>>>(partial, log_v);
    }
    k_rowmax<<<N, 256, 0, stream>>>(out, log_u, log_v, rowmax);
    k_gmax<<<1, 256, 0, stream>>>(rowmax, gmax);
    k_final<<<N, 256, 0, stream>>>(out, log_u, log_v, gmax);
}
